// GCN_32289564131895
// MI455X (gfx1250) — matrix operand order repaired and checked by reading, not compiled, not run on hardware
//
#include <hip/hip_runtime.h>
#include <hip/hip_bf16.h>
#include <math.h>

// ---------------------------------------------------------------------------
// Types for CDNA5 WMMA (wave32): 16x16x32 bf16 -> f32
// ---------------------------------------------------------------------------
typedef __attribute__((ext_vector_type(16))) __bf16 v16bf;
typedef __attribute__((ext_vector_type(8)))  float  v8f;
typedef __attribute__((ext_vector_type(4)))  unsigned int u32x4;

union FragB16 { v16bf v; u32x4 q[2]; };

// Problem constants
#define NQ     4096
#define C1N    1001
#define DIM    4096
#define HMLP   1024
#define HH     256
#define OO     256
#define TOPK   32
#define K1P    5120   // 5101 padded to multiple of 64

// LDS tile geometry for the WMMA GEMM
#define TM     128    // block rows   (8 waves x 16)
#define TN     64     // block cols   (4 wmma tiles per wave)
#define TK     64     // K per stage  (2 wmma k-chunks)
#define LDP    72     // padded row length in bf16 (+16B to spread LDS banks)

// ---------------------------------------------------------------------------
// gfx1250 async copy global->LDS (ASYNCcnt-tracked, VFLAT encoding)
// ---------------------------------------------------------------------------
__device__ __forceinline__ unsigned ldsAddr(const void* p) {
    // Low 32 bits of a flat pointer into the LDS aperture are the LDS offset.
    return (unsigned)(unsigned long long)(uintptr_t)p;
}

__device__ __forceinline__ void async_copy_b128(unsigned lds, const void* g) {
    asm volatile("global_load_async_to_lds_b128 %0, %1, off"
                 :: "v"(lds), "v"((unsigned long long)(uintptr_t)g)
                 : "memory");
}

__device__ __forceinline__ void wait_async_le6() {
    asm volatile("s_wait_asynccnt 0x6" ::: "memory");
}
__device__ __forceinline__ void wait_async_le0() {
    asm volatile("s_wait_asynccnt 0x0" ::: "memory");
}

// ---------------------------------------------------------------------------
// Pack ew_in = concat(query_emb, probs, bbox) -> bf16 [N][5120] (zero padded)
// ---------------------------------------------------------------------------
__global__ __launch_bounds__(256)
void pack_ewin(const float* __restrict__ qe, const float* __restrict__ probs,
               const float* __restrict__ bbox, __bf16* __restrict__ A, int N)
{
    size_t total = (size_t)N * K1P;
    for (size_t i = (size_t)blockIdx.x * blockDim.x + threadIdx.x; i < total;
         i += (size_t)gridDim.x * blockDim.x) {
        int n = (int)(i / K1P);
        int k = (int)(i % K1P);
        float v;
        if (k < DIM)                v = qe[(size_t)n * DIM + k];
        else if (k < DIM + C1N)     v = probs[(size_t)n * C1N + (k - DIM)];
        else if (k < DIM + C1N + 4) v = bbox[(size_t)n * 4 + (k - DIM - C1N)];
        else                        v = 0.0f;
        A[i] = (__bf16)v;
    }
}

// ---------------------------------------------------------------------------
// Transpose + convert weight W[K][Nn] (f32, row-major) -> Bt[Nn][Kp] (bf16)
// ---------------------------------------------------------------------------
__global__ __launch_bounds__(256)
void conv_transpose_bf16(const float* __restrict__ W, __bf16* __restrict__ Bt,
                         int K, int Nn, int Kp)
{
    size_t total = (size_t)Nn * Kp;
    for (size_t i = (size_t)blockIdx.x * blockDim.x + threadIdx.x; i < total;
         i += (size_t)gridDim.x * blockDim.x) {
        int n = (int)(i / Kp);
        int k = (int)(i % Kp);
        Bt[i] = (__bf16)((k < K) ? W[(size_t)k * Nn + n] : 0.0f);
    }
}

// ---------------------------------------------------------------------------
// WMMA GEMM: C[M][Nn] = A[M][K](bf16) x Bt[Nn][K](bf16, pre-transposed) + bias
//
// Block = 256 threads = 8 waves; block tile 128x64; K staged through LDS in
// steps of 64 with DOUBLE BUFFERING driven by global_load_async_to_lds_b128:
//   main loop: issue(stage s+1) -> s_wait_asynccnt 6 (stage s complete,
//   s+1 in flight) -> barrier -> 20 ds_load_b128 then 8 v_wmma (in-order DS
//   returns allow partial s_wait_dscnt gating) -> barrier.  Epilogue peeled.
// Fragment layouts per CDNA5 ISA:
//   A (16-bit 16x32): lane row = lane&15, K = (lane>>4)*8 + {0..7, 16..23}
//   B (16-bit 32x16): lane col = lane&15, K = (lane>>4)*8 + {0..7, 16..23}
//   C/D: M = v + (lane>>4)*8, N = lane&15 for acc element v.
// Requires M%128==0, Nn%64==0, K%128==0 at call sites (>=2 stages).
// ---------------------------------------------------------------------------
__global__ __launch_bounds__(256)
void wmma_gemm_bf16(const __bf16* __restrict__ A, const __bf16* __restrict__ Bt,
                    const float* __restrict__ bias,
                    float* __restrict__ Cf, __bf16* __restrict__ Cbf,
                    int M, int Nn, int K, int relu)
{
    __shared__ __attribute__((aligned(16))) __bf16 sA[2][TM][LDP];
    __shared__ __attribute__((aligned(16))) __bf16 sB[2][TN][LDP];

    const int tid  = threadIdx.x;
    const int lane = tid & 31;
    const int wave = tid >> 5;                 // 0..7
    const int half = lane >> 4;                // 0/1
    const int l16  = lane & 15;

    const int rowM    = blockIdx.y * TM;       // block row base
    const int colBase = blockIdx.x * TN;       // block col base

    // ---- async stage issue: 4 A-chunks + 2 B-chunks of 16B per thread ----
    auto issueStage = [&](int k0, int buf) {
#pragma unroll
        for (int j = 0; j < 4; ++j) {          // A tile: 128x64 bf16 = 16KB
            int c   = tid + j * 256;           // 1024 chunks of 16B
            int row = c >> 3;
            int cc  = c & 7;
            const __bf16* g = A + (size_t)(rowM + row) * K + k0 + cc * 8;
            async_copy_b128(ldsAddr(&sA[buf][row][cc * 8]), g);
        }
#pragma unroll
        for (int j = 0; j < 2; ++j) {          // B tile: 64x64 bf16 = 8KB
            int c   = tid + j * 256;           // 512 chunks of 16B
            int col = c >> 3;
            int kc  = c & 7;
            const __bf16* g = Bt + (size_t)(colBase + col) * K + k0 + kc * 8;
            async_copy_b128(ldsAddr(&sB[buf][col][kc * 8]), g);
        }
    };

    // ---- accumulators (bias broadcast: column = colBase + t*16 + l16) ----
    v8f acc[4];
#pragma unroll
    for (int t = 0; t < 4; ++t) {
        float b = bias ? bias[colBase + t * 16 + l16] : 0.0f;
#pragma unroll
        for (int v = 0; v < 8; ++v) acc[t][v] = b;
    }

    // ---- compute one LDS-resident stage: batch 20 ds_load_b128, 8 wmma ----
    auto computeStage = [&](int buf) {
        FragB16 fa[2];
        FragB16 fb[2][4];
        const __bf16* aRow = &sA[buf][wave * 16 + l16][0];
        const __bf16* bRow = &sB[buf][l16][0];
#pragma unroll
        for (int h = 0; h < 2; ++h) {          // k-chunk kk = h*32
            const __bf16* ap = aRow + h * 32 + half * 8;
            fa[h].q[0] = *(const u32x4*)(ap);        // K = base+0..7
            fa[h].q[1] = *(const u32x4*)(ap + 16);   // K = base+16..23
#pragma unroll
            for (int t = 0; t < 4; ++t) {
                const __bf16* bp = bRow + (size_t)t * 16 * LDP + h * 32 + half * 8;
                fb[h][t].q[0] = *(const u32x4*)(bp);      // K = base+0..7
                fb[h][t].q[1] = *(const u32x4*)(bp + 16); // K = base+16..23
            }
        }
#pragma unroll
        for (int h = 0; h < 2; ++h)
#pragma unroll
            for (int t = 0; t < 4; ++t)
                acc[t] = __builtin_amdgcn_wmma_f32_16x16x32_bf16(
                    false, fa[h].v, false, fb[h][t].v, (short)0, acc[t],
                    false, false);
    };

    const int nStages = K / TK;
    issueStage(0, 0);

    int s = 0;
    for (; s < nStages - 1; ++s) {             // main loop: always prefetch
        issueStage((s + 1) * TK, (s & 1) ^ 1);
        wait_async_le6();                      // stage s done, s+1 in flight
        __syncthreads();
        computeStage(s & 1);
        __syncthreads();                       // buf reusable next iteration
    }
    wait_async_le0();                          // peeled epilogue
    __syncthreads();
    computeStage(s & 1);

    // ---- store (C/D layout) ----
    const int mrow = rowM + wave * 16 + half * 8;
#pragma unroll
    for (int t = 0; t < 4; ++t) {
        int col = colBase + t * 16 + l16;
#pragma unroll
        for (int v = 0; v < 8; ++v) {
            float x = acc[t][v];
            if (relu) x = fmaxf(x, 0.0f);
            size_t o = (size_t)(mrow + v) * Nn + col;
            if (Cf)  Cf[o]  = x;
            if (Cbf) Cbf[o] = (__bf16)x;
        }
    }
}

// ---------------------------------------------------------------------------
// Top-32 per row of adj (N x N) + degree -> dinv. One block (256 thr) / row.
// Row staged in LDS; 32 rounds of LDS argmax reduction (tie -> lowest index).
// ---------------------------------------------------------------------------
__global__ __launch_bounds__(256)
void topk_rows(const float* __restrict__ adj, float* __restrict__ vals,
               int* __restrict__ idx, float* __restrict__ dinv, int N)
{
    __shared__ float srow[NQ];
    __shared__ float sv[256];
    __shared__ int   si[256];

    const int row = blockIdx.x;
    const int tid = threadIdx.x;
    const float* arow = adj + (size_t)row * N;

    for (int j = tid; j < N; j += 256) srow[j] = arow[j];
    __syncthreads();

    float deg = 0.0f;
    for (int t = 0; t < TOPK; ++t) {
        float best = -INFINITY; int bidx = 0x7fffffff;
        for (int j = tid; j < N; j += 256) {
            float v = srow[j];
            if (v > best) { best = v; bidx = j; }
        }
        sv[tid] = best; si[tid] = bidx;
        __syncthreads();
        for (int s = 128; s > 0; s >>= 1) {
            if (tid < s) {
                float ov = sv[tid + s]; int oi = si[tid + s];
                if (ov > sv[tid] || (ov == sv[tid] && oi < si[tid])) {
                    sv[tid] = ov; si[tid] = oi;
                }
            }
            __syncthreads();
        }
        if (tid == 0) {
            vals[(size_t)row * TOPK + t] = sv[0];
            idx[(size_t)row * TOPK + t]  = si[0];
            deg += sv[0];
            srow[si[0]] = -INFINITY;
        }
        __syncthreads();
    }
    if (tid == 0)
        dinv[row] = (deg > 0.0f) ? rsqrtf(fmaxf(deg, 1e-12f)) : 0.0f;
}

// ---------------------------------------------------------------------------
// Fused wn + gather-aggregate: out[n][f] = sum_k wn[n][k] * X[idx[n][k]][f]
// wn[n][k] = vals[n][k] * dinv[n] * dinv[idx[n][k]].  Output bf16 (GEMM A).
// ---------------------------------------------------------------------------
__global__ __launch_bounds__(256)
void wn_gather_bf16(const float* __restrict__ vals, const int* __restrict__ idx,
                    const float* __restrict__ dinv, const float* __restrict__ X,
                    __bf16* __restrict__ out, int F)
{
    __shared__ float swn[TOPK];
    __shared__ int   sidx[TOPK];
    const int row = blockIdx.x;
    const int tid = threadIdx.x;

    if (tid < TOPK) {
        int j = idx[(size_t)row * TOPK + tid];
        sidx[tid] = j;
        swn[tid] = vals[(size_t)row * TOPK + tid] * dinv[row] * dinv[j];
    }
    __syncthreads();

    for (int f = tid; f < F; f += 256) {
        float acc = 0.0f;
#pragma unroll 8
        for (int k = 0; k < TOPK; ++k)
            acc += swn[k] * X[(size_t)sidx[k] * F + f];
        out[(size_t)row * F + f] = (__bf16)acc;
    }
}

// ---------------------------------------------------------------------------
// BatchNorm column statistics: one block per feature (F=256).
// ---------------------------------------------------------------------------
__global__ __launch_bounds__(256)
void bn_stats(const float* __restrict__ X, float* __restrict__ mean,
              float* __restrict__ varr, int N, int F)
{
    __shared__ float ss[256], sq[256];
    const int f = blockIdx.x;
    const int tid = threadIdx.x;
    float s = 0.0f, q = 0.0f;
    for (int n = tid; n < N; n += 256) {
        float v = X[(size_t)n * F + f];
        s += v; q += v * v;
    }
    ss[tid] = s; sq[tid] = q;
    __syncthreads();
    for (int st = 128; st > 0; st >>= 1) {
        if (tid < st) { ss[tid] += ss[tid + st]; sq[tid] += sq[tid + st]; }
        __syncthreads();
    }
    if (tid == 0) {
        float m = ss[0] / (float)N;
        mean[f] = m;
        varr[f] = sq[0] / (float)N - m * m;
    }
}

__global__ __launch_bounds__(256)
void bn_apply_relu(const float* __restrict__ X, const float* __restrict__ mean,
                   const float* __restrict__ varr, const float* __restrict__ gamma,
                   const float* __restrict__ beta, float* __restrict__ Y,
                   int N, int F)
{
    size_t total = (size_t)N * F;
    for (size_t i = (size_t)blockIdx.x * blockDim.x + threadIdx.x; i < total;
         i += (size_t)gridDim.x * blockDim.x) {
        int f = (int)(i % F);
        float v = (X[i] - mean[f]) * rsqrtf(varr[f] + 1e-5f) * gamma[f] + beta[f];
        Y[i] = fmaxf(v, 0.0f);
    }
}

// ---------------------------------------------------------------------------
// Host launcher
// ---------------------------------------------------------------------------
extern "C" void kernel_launch(void* const* d_in, const int* in_sizes, int n_in,
                              void* d_out, int out_size, void* d_ws, size_t ws_size,
                              hipStream_t stream)
{
    const float* probs   = (const float*)d_in[0];
    const float* bbox    = (const float*)d_in[1];
    const float* qemb    = (const float*)d_in[2];
    const float* nemb    = (const float*)d_in[3];
    const float* mlp_w1  = (const float*)d_in[4];
    const float* mlp_b1  = (const float*)d_in[5];
    const float* mlp_w2  = (const float*)d_in[6];
    const float* mlp_b2  = (const float*)d_in[7];
    const float* conv_w1 = (const float*)d_in[8];
    const float* conv_b1 = (const float*)d_in[9];
    const float* conv_w2 = (const float*)d_in[10];
    const float* conv_b2 = (const float*)d_in[11];
    const float* bn_g    = (const float*)d_in[12];
    const float* bn_b    = (const float*)d_in[13];
    float* out = (float*)d_out;

    char* ws = (char*)d_ws;
    // Zone (0..67MB) is reused: {A1,Bt1} -> adj -> {msg1,out1,y,msg2}
    __bf16* A1   = (__bf16*)(ws + 0);                       // 4096x5120 bf16
    __bf16* Bt1  = (__bf16*)(ws + 41943040);                // 1024x5120 bf16
    float*  adj  = (float*) (ws + 0);                       // 4096x4096 f32
    __bf16* msg1 = (__bf16*)(ws + 0);                       // 4096x4096 bf16
    float*  out1 = (float*) (ws + 33554432);                // 4096x256  f32
    float*  ybn  = (float*) (ws + 37748736);                // 4096x256  f32
    __bf16* msg2 = (__bf16*)(ws + 41943040);                // 4096x256  bf16
    __bf16* Hbf  = (__bf16*)(ws + 67108864);                // 4096x1024 bf16
    __bf16* Bt2  = (__bf16*)(ws + 75497472);                // 4096x1024 bf16
    float*  vals = (float*) (ws + 83886080);                // 4096x32
    int*    idxb = (int*)   (ws + 84410368);                // 4096x32
    float*  dinv = (float*) (ws + 84934656);                // 4096
    __bf16* Btc1 = (__bf16*)(ws + 84951040);                // 256x4096 bf16
    __bf16* Btc2 = (__bf16*)(ws + 87048192);                // 256x256  bf16
    float*  bmean= (float*) (ws + 87179264);                // 256
    float*  bvar = (float*) (ws + 87180288);                // 256
    (void)ws_size; (void)n_in; (void)in_sizes; (void)out_size;

    dim3 blk(256);

    // 1) pack ew_in (bf16, K padded to 5120)
    pack_ewin<<<8192, blk, 0, stream>>>(qemb, probs, bbox, A1, NQ);
    // 2) W1^T -> bf16 [1024][5120]
    conv_transpose_bf16<<<8192, blk, 0, stream>>>(mlp_w1, Bt1, DIM + C1N + 4, HMLP, K1P);
    // 3) H = relu(ew_in @ W1 + b1) -> bf16 [4096][1024]
    wmma_gemm_bf16<<<dim3(HMLP / TN, NQ / TM), blk, 0, stream>>>(
        A1, Bt1, mlp_b1, nullptr, Hbf, NQ, HMLP, K1P, 1);
    // 4) W2^T -> bf16 [4096][1024]
    conv_transpose_bf16<<<8192, blk, 0, stream>>>(mlp_w2, Bt2, HMLP, NQ, HMLP);
    // 5) adj = H @ W2 + b2 -> f32 [4096][4096]  (overwrites A1/Bt1 zone)
    wmma_gemm_bf16<<<dim3(NQ / TN, NQ / TM), blk, 0, stream>>>(
        Hbf, Bt2, mlp_b2, adj, nullptr, NQ, NQ, HMLP, 0);
    // 6) top-32 per row + dinv
    topk_rows<<<NQ, blk, 0, stream>>>(adj, vals, idxb, dinv, NQ);
    // 7) conv_w1^T -> bf16 [256][4096]
    conv_transpose_bf16<<<4096, blk, 0, stream>>>(conv_w1, Btc1, DIM, HH, DIM);
    // 8) msg1 = gather-agg(node_emb) -> bf16 [4096][4096] (reuses adj zone)
    wn_gather_bf16<<<NQ, blk, 0, stream>>>(vals, idxb, dinv, nemb, msg1, DIM);
    // 9) out1 = msg1 @ conv_w1 + b -> f32 [4096][256]
    wmma_gemm_bf16<<<dim3(HH / TN, NQ / TM), blk, 0, stream>>>(
        msg1, Btc1, conv_b1, out1, nullptr, NQ, HH, DIM, 0);
    // 10) batch-norm statistics
    bn_stats<<<HH, blk, 0, stream>>>(out1, bmean, bvar, NQ, HH);
    // 11) y = relu(BN(out1))
    bn_apply_relu<<<4096, blk, 0, stream>>>(out1, bmean, bvar, bn_g, bn_b, ybn, NQ, HH);
    // 12) msg2 = gather-agg(y) -> bf16 [4096][256]
    wn_gather_bf16<<<NQ, blk, 0, stream>>>(vals, idxb, dinv, ybn, msg2, HH);
    // 13) conv_w2^T -> bf16 [256][256]
    conv_transpose_bf16<<<256, blk, 0, stream>>>(conv_w2, Btc2, HH, OO, HH);
    // 14) out = msg2 @ conv_w2 + b2 -> f32 [4096][256]
    wmma_gemm_bf16<<<dim3(OO / TN, NQ / TM), blk, 0, stream>>>(
        msg2, Btc2, conv_b2, out, nullptr, NQ, OO, HH, 0);
}
